// Block_45578192945855
// MI455X (gfx1250) — matrix operand order repaired and checked by reading, not compiled, not run on hardware
//
#include <hip/hip_runtime.h>
#include <hip/hip_bf16.h>

#define D_DIM 2048
#define T_DIM 8192
#define WKV_L 128                 // scan chunk length
#define WKV_C (T_DIM / WKV_L)     // 64 chunks

typedef __attribute__((ext_vector_type(16))) __bf16 v16bf;
typedef __attribute__((ext_vector_type(8)))  __bf16 v8bf;
typedef __attribute__((ext_vector_type(8)))  float  v8f;

#define BM 128
#define BN 128
#define BK 32
#define LDA 40   // padded LDS row stride (bf16): 80B rows, 16B-aligned segments

// ---------------------------------------------------------------------------
// CDNA5 async global->LDS copy (ASYNCcnt-tracked), one 16B chunk per lane.
// ---------------------------------------------------------------------------
__device__ __forceinline__ void async_copy_b128(uint32_t lds_byte_addr,
                                                const void* gaddr)
{
    asm volatile("global_load_async_to_lds_b128 %0, %1, off"
                 :: "v"(lds_byte_addr), "v"(gaddr) : "memory");
}
__device__ __forceinline__ void wait_asynccnt0()
{
    asm volatile("s_wait_asynccnt 0" ::: "memory");
}

// ---------------------------------------------------------------------------
// One-time f32 -> bf16 weight conversion (weights are static per launch).
// ---------------------------------------------------------------------------
__global__ __launch_bounds__(256)
void wcvt_kernel(const float* __restrict__ src, __bf16* __restrict__ dst)
{
    const size_t i = ((size_t)blockIdx.x * 256 + threadIdx.x) * 4;
    const float4 c = *(const float4*)(src + i);
    dst[i + 0] = (__bf16)c.x;
    dst[i + 1] = (__bf16)c.y;
    dst[i + 2] = (__bf16)c.z;
    dst[i + 3] = (__bf16)c.w;
}

// ---------------------------------------------------------------------------
// LayerNorm: one block per row, 256 threads, D=2048.
// ---------------------------------------------------------------------------
__global__ __launch_bounds__(256)
void ln_kernel(const float* __restrict__ x, const float* __restrict__ g,
               const float* __restrict__ b, float* __restrict__ out)
{
    const int row = blockIdx.x;
    const int tid = threadIdx.x;
    const float* xr = x + (size_t)row * D_DIM;

    float s = 0.f, s2 = 0.f;
    #pragma unroll
    for (int i = tid; i < D_DIM; i += 256) {
        float v = xr[i];
        s += v; s2 += v * v;
    }
    __shared__ float red[512];
    red[tid] = s; red[256 + tid] = s2;
    __syncthreads();
    for (int off = 128; off > 0; off >>= 1) {
        if (tid < off) {
            red[tid]       += red[tid + off];
            red[256 + tid] += red[256 + tid + off];
        }
        __syncthreads();
    }
    const float mean = red[0]   * (1.f / D_DIM);
    const float var  = red[256] * (1.f / D_DIM) - mean * mean;
    const float inv  = rsqrtf(var + 1e-5f);
    float* orow = out + (size_t)row * D_DIM;
    #pragma unroll
    for (int i = tid; i < D_DIM; i += 256)
        orow[i] = (xr[i] - mean) * inv * g[i] + b[i];
}

// ---------------------------------------------------------------------------
// WKV chunk-parallel scan, 3 passes.  State (aa,bb,pp) represents
// A = aa*e^pp, B = bb*e^pp; the per-chunk transform is affine with decay
// e^{L*w}, so chunks combine with the same stable log-sum-exp merge.
// ---------------------------------------------------------------------------

// Pass 1: per (chunk, channel), reduce chunk from empty state.
__global__ __launch_bounds__(256)
void wkv_local_kernel(const float* __restrict__ K, const float* __restrict__ V,
                      const float* __restrict__ td,
                      float* __restrict__ Saa, float* __restrict__ Sbb,
                      float* __restrict__ Spp)
{
    const int gid = blockIdx.x * 256 + threadIdx.x;   // c*D + d
    const int d = gid & (D_DIM - 1);
    const int c = gid >> 11;
    const float w = -__expf(td[d]);
    float aa = 0.f, bb = 0.f, pp = -1e38f;
    const size_t base = (size_t)c * WKV_L * D_DIM + d;
    for (int i = 0; i < WKV_L; ++i) {
        const size_t idx = base + (size_t)i * D_DIM;
        const float kt = K[idx];
        const float vt = V[idx];
        const float ww2 = pp + w;
        const float p2  = fmaxf(ww2, kt);
        const float e1  = __expf(ww2 - p2);
        const float e2  = __expf(kt - p2);
        aa = e1 * aa + e2 * vt;
        bb = e1 * bb + e2;
        pp = p2;
    }
    Saa[gid] = aa; Sbb[gid] = bb; Spp[gid] = pp;
}

// Pass 2: per channel, exclusive prefix combine over the 64 chunk states.
__global__ __launch_bounds__(256)
void wkv_prefix_kernel(const float* __restrict__ td,
                       float* __restrict__ Saa, float* __restrict__ Sbb,
                       float* __restrict__ Spp)
{
    const int d = blockIdx.x * 256 + threadIdx.x;
    if (d >= D_DIM) return;
    const float w  = -__expf(td[d]);
    const float Lw = (float)WKV_L * w;
    float aa = 0.f, bb = 0.f, pp = -1e38f;
    for (int c = 0; c < WKV_C; ++c) {
        const int idx = c * D_DIM + d;
        const float la = Saa[idx], lb = Sbb[idx], lp = Spp[idx];
        Saa[idx] = aa; Sbb[idx] = bb; Spp[idx] = pp;   // exclusive prefix
        const float q  = pp + Lw;                      // decay prefix by L steps
        const float p  = fmaxf(q, lp);
        const float e1 = __expf(q - p);
        const float e2 = __expf(lp - p);
        aa = e1 * aa + e2 * la;
        bb = e1 * bb + e2 * lb;
        pp = p;
    }
}

// Pass 3: replay each chunk from its prefix state, emit wkv*sigmoid(r).
// OUT may alias K (same element read-then-written by the same thread).
__global__ __launch_bounds__(256)
void wkv_apply_kernel(const float* __restrict__ K, const float* __restrict__ V,
                      const float* __restrict__ R, const float* __restrict__ tf,
                      const float* __restrict__ td,
                      const float* __restrict__ Saa, const float* __restrict__ Sbb,
                      const float* __restrict__ Spp, float* __restrict__ OUT)
{
    const int gid = blockIdx.x * 256 + threadIdx.x;   // c*D + d
    const int d = gid & (D_DIM - 1);
    const int c = gid >> 11;
    const float u = tf[d];
    const float w = -__expf(td[d]);
    float aa = Saa[gid], bb = Sbb[gid], pp = Spp[gid];
    const size_t base = (size_t)c * WKV_L * D_DIM + d;
    for (int i = 0; i < WKV_L; ++i) {
        const size_t idx = base + (size_t)i * D_DIM;
        const float kt = K[idx];
        const float vt = V[idx];
        const float rt = R[idx];
        float ww = u + kt;
        float p  = fmaxf(pp, ww);
        float e1 = __expf(pp - p);
        float e2 = __expf(ww - p);
        OUT[idx] = ((e1 * aa + e2 * vt) / (e1 * bb + e2)) * rt;
        float ww2 = pp + w;
        float p2  = fmaxf(ww2, kt);
        e1 = __expf(ww2 - p2);
        e2 = __expf(kt - p2);
        aa = e1 * aa + e2 * vt;
        bb = e1 * bb + e2;
        pp = p2;
    }
}

// ---------------------------------------------------------------------------
// Fused GEMM: out = epi(Astage @ Wb^T + bias) [* mulp] [+ addp]
//   AMODE 0: Astage = A0 (f32, cvt to bf16 in staging)
//   AMODE 1: Astage = lerp(roll(A0,1), A0, mix)  (token shift, fused)
//   EPI   0: identity   1: sigmoid   2: relu then square
// Wb is pre-converted bf16; B tiles move global->LDS with
// global_load_async_to_lds_b128, double-buffered against the WMMA stage.
// 128x128 block tile, BK=32, 8 waves x (4x2) v_wmma_f32_16x16x32_bf16.
// ---------------------------------------------------------------------------
template<int AMODE, int EPI, int HASMUL, int HASADD>
__global__ __launch_bounds__(256)
void gemm_rwkv(const float* __restrict__ A0, const float* __restrict__ mix,
               const __bf16* __restrict__ Wb, const float* __restrict__ bias,
               const float* __restrict__ mulp, const float* __restrict__ addp,
               float* __restrict__ out)
{
    constexpr int Kd = D_DIM, Nd = D_DIM;
    __shared__ alignas(16) __bf16 As[2][BM][LDA];
    __shared__ alignas(16) __bf16 Bs[2][BN][LDA];

    const int tid  = threadIdx.x;
    const int lane = tid & 31;
    const int wave = tid >> 5;
    const int wm   = wave >> 2;          // 0..1 (64 rows)
    const int wn   = wave & 3;           // 0..3 (32 cols)
    const int m0   = blockIdx.y * BM;
    const int n0   = blockIdx.x * BN;

    // A staging: 2 threads per tile row, 16 f32 each (fused lerp + cvt)
    const int sr = tid >> 1;
    const int sc = (tid & 1) * 16;
    const int tg  = m0 + sr;
    const int tgp = (tg + T_DIM - 1) % T_DIM;

    // B staging: 2 async b128 per lane; chunk = wave*64 + j*32 + lane
    const int chunk0 = wave * 64 + lane;
    const int chunk1 = chunk0 + 32;
    const int brow0 = chunk0 >> 2, bseg0 = chunk0 & 3;
    const int brow1 = chunk1 >> 2, bseg1 = chunk1 & 3;

    auto stage = [&](int buf, int k0) {
        // ---- B tile: pure async copy, no VALU ----
        async_copy_b128((uint32_t)(uintptr_t)&Bs[buf][brow0][bseg0 * 8],
                        Wb + (size_t)(n0 + brow0) * Kd + k0 + bseg0 * 8);
        async_copy_b128((uint32_t)(uintptr_t)&Bs[buf][brow1][bseg1 * 8],
                        Wb + (size_t)(n0 + brow1) * Kd + k0 + bseg1 * 8);
        // ---- A tile: fused token-shift lerp + f32->bf16 ----
        const float4* curp = (const float4*)(A0 + (size_t)tg * Kd + k0 + sc);
        if (AMODE == 1) {
            const float4* prvp = (const float4*)(A0 + (size_t)tgp * Kd + k0 + sc);
            const float4* mixp = (const float4*)(mix + k0 + sc);
            #pragma unroll
            for (int i = 0; i < 4; ++i) {
                float4 c = curp[i], p = prvp[i], m = mixp[i];
                As[buf][sr][sc + 4*i + 0] = (__bf16)(p.x + m.x * (c.x - p.x));
                As[buf][sr][sc + 4*i + 1] = (__bf16)(p.y + m.y * (c.y - p.y));
                As[buf][sr][sc + 4*i + 2] = (__bf16)(p.z + m.z * (c.z - p.z));
                As[buf][sr][sc + 4*i + 3] = (__bf16)(p.w + m.w * (c.w - p.w));
            }
        } else {
            #pragma unroll
            for (int i = 0; i < 4; ++i) {
                float4 c = curp[i];
                As[buf][sr][sc + 4*i + 0] = (__bf16)c.x;
                As[buf][sr][sc + 4*i + 1] = (__bf16)c.y;
                As[buf][sr][sc + 4*i + 2] = (__bf16)c.z;
                As[buf][sr][sc + 4*i + 3] = (__bf16)c.w;
            }
        }
    };

    v8f acc[4][2];
    const v8f zero = {0.f,0.f,0.f,0.f,0.f,0.f,0.f,0.f};
    #pragma unroll
    for (int mi = 0; mi < 4; ++mi)
        #pragma unroll
        for (int ni = 0; ni < 2; ++ni) acc[mi][ni] = zero;

    // fragment coordinates per ISA 16-bit WMMA layouts
    const int ma = wm * 64 + (lane & 15);
    const int kh = (lane >> 4) * 8;
    const int nb = wn * 32 + (lane & 15);
    const int kb = (lane >> 4) * 16;

    stage(0, 0);
    for (int k0 = 0, it = 0; k0 < Kd; k0 += BK, ++it) {
        const int buf = it & 1;
        wait_asynccnt0();        // own async stores to LDS complete
        __syncthreads();         // everyone's staging visible; prev compute done
        if (k0 + BK < Kd) stage(buf ^ 1, k0 + BK);  // overlap with WMMAs below

        v16bf af[4], bfr[2];
        #pragma unroll
        for (int mi = 0; mi < 4; ++mi) {
            const v8bf lo = *(const v8bf*)&As[buf][ma + mi*16][kh];
            const v8bf hi = *(const v8bf*)&As[buf][ma + mi*16][16 + kh];
            #pragma unroll
            for (int e = 0; e < 8; ++e) { af[mi][e] = lo[e]; af[mi][8+e] = hi[e]; }
        }
        #pragma unroll
        for (int ni = 0; ni < 2; ++ni) {
            const v8bf lo = *(const v8bf*)&Bs[buf][nb + ni*16][kh];
            const v8bf hi = *(const v8bf*)&Bs[buf][nb + ni*16][16 + kh];
            #pragma unroll
            for (int e = 0; e < 8; ++e) { bfr[ni][e] = lo[e]; bfr[ni][8+e] = hi[e]; }
        }
        #pragma unroll
        for (int mi = 0; mi < 4; ++mi)
            #pragma unroll
            for (int ni = 0; ni < 2; ++ni)
                acc[mi][ni] = __builtin_amdgcn_wmma_f32_16x16x32_bf16(
                    false, af[mi], false, bfr[ni], (short)0, acc[mi][ni],
                    false, false);
    }

    // ---- epilogue: bias, activation, compile-time gate/residual ----
    const int colb = n0 + wn * 32 + (lane & 15);
    const int rowb = m0 + wm * 64 + (lane >> 4) * 8;  // C/D: M = j + 8*(lane/16)
    #pragma unroll
    for (int ni = 0; ni < 2; ++ni) {
        const int col = colb + ni * 16;
        const float bv = bias[col];
        #pragma unroll
        for (int mi = 0; mi < 4; ++mi) {
            #pragma unroll
            for (int j = 0; j < 8; ++j) {
                const int row = rowb + mi * 16 + j;
                float v = acc[mi][ni][j] + bv;
                if (EPI == 1)      v = 1.f / (1.f + __expf(-v));
                else if (EPI == 2) { v = fmaxf(v, 0.f); v = v * v; }
                const size_t idx = (size_t)row * Nd + col;
                if (HASMUL) v *= mulp[idx];
                if (HASADD) v += addp[idx];
                out[idx] = v;
            }
        }
    }
}

// ---------------------------------------------------------------------------
extern "C" void kernel_launch(void* const* d_in, const int* in_sizes, int n_in,
                              void* d_out, int out_size, void* d_ws, size_t ws_size,
                              hipStream_t stream)
{
    const float* x       = (const float*)d_in[0];
    const float* attk_w  = (const float*)d_in[1];
    const float* attk_b  = (const float*)d_in[2];
    const float* attv_w  = (const float*)d_in[3];
    const float* attv_b  = (const float*)d_in[4];
    const float* attr_w  = (const float*)d_in[5];
    const float* attr_b  = (const float*)d_in[6];
    const float* atto_w  = (const float*)d_in[7];
    const float* atto_b  = (const float*)d_in[8];
    const float* ffnk_w  = (const float*)d_in[9];
    const float* ffnk_b  = (const float*)d_in[10];
    const float* ffnv_w  = (const float*)d_in[11];
    const float* ffnv_b  = (const float*)d_in[12];
    const float* ffnr_w  = (const float*)d_in[13];
    const float* ffnr_b  = (const float*)d_in[14];
    const float* ln1_g   = (const float*)d_in[15];
    const float* ln1_b   = (const float*)d_in[16];
    const float* ln2_g   = (const float*)d_in[17];
    const float* ln2_b   = (const float*)d_in[18];
    const float* attmixk = (const float*)d_in[19];
    const float* attmixv = (const float*)d_in[20];
    const float* attmixr = (const float*)d_in[21];
    const float* ffnmixk = (const float*)d_in[22];
    const float* ffnmixr = (const float*)d_in[23];
    const float* tfirst  = (const float*)d_in[24];
    const float* tdecay  = (const float*)d_in[25];

    const size_t TD = (size_t)T_DIM * D_DIM;
    const size_t WD = (size_t)D_DIM * D_DIM;
    const size_t CD = (size_t)WKV_C * D_DIM;
    float* xy = (float*)d_ws;        // xy, later ddd
    float* kb = xy + TD;             // k, then wkv*r, then relu(kf)^2
    float* vb = xy + 2 * TD;         // v, then rz
    float* rb = xy + 3 * TD;         // sigmoid(r), then rf
    __bf16* wbf = (__bf16*)(xy + 4 * TD);   // 7 bf16 weight matrices
    __bf16* w_attk = wbf + 0 * WD;
    __bf16* w_attv = wbf + 1 * WD;
    __bf16* w_attr = wbf + 2 * WD;
    __bf16* w_atto = wbf + 3 * WD;
    __bf16* w_ffnk = wbf + 4 * WD;
    __bf16* w_ffnv = wbf + 5 * WD;
    __bf16* w_ffnr = wbf + 6 * WD;
    float* scanws = (float*)(wbf + 7 * WD); // chunk states: 3 x [C][D]
    float* Saa = scanws;
    float* Sbb = scanws + CD;
    float* Spp = scanws + 2 * CD;
    float* outp = (float*)d_out;

    const dim3 blk(256);
    const dim3 grd(D_DIM / BN, T_DIM / BM);
    const int cvtg  = (int)(WD / 1024);          // wcvt: 4 elems/thread
    const int scang = (int)(CD / 256);           // 512 blocks, thread per (c,d)

    // --- one-time weight conversion to bf16 (L2-resident) ---
    wcvt_kernel<<<cvtg, blk, 0, stream>>>(attk_w, w_attk);
    wcvt_kernel<<<cvtg, blk, 0, stream>>>(attv_w, w_attv);
    wcvt_kernel<<<cvtg, blk, 0, stream>>>(attr_w, w_attr);
    wcvt_kernel<<<cvtg, blk, 0, stream>>>(atto_w, w_atto);
    wcvt_kernel<<<cvtg, blk, 0, stream>>>(ffnk_w, w_ffnk);
    wcvt_kernel<<<cvtg, blk, 0, stream>>>(ffnv_w, w_ffnv);
    wcvt_kernel<<<cvtg, blk, 0, stream>>>(ffnr_w, w_ffnr);

    // --- time-mix branch ---
    ln_kernel<<<T_DIM, blk, 0, stream>>>(x, ln1_g, ln1_b, xy);
    gemm_rwkv<1,0,0,0><<<grd, blk, 0, stream>>>(xy, attmixk, w_attk, attk_b, nullptr, nullptr, kb);
    gemm_rwkv<1,0,0,0><<<grd, blk, 0, stream>>>(xy, attmixv, w_attv, attv_b, nullptr, nullptr, vb);
    gemm_rwkv<1,1,0,0><<<grd, blk, 0, stream>>>(xy, attmixr, w_attr, attr_b, nullptr, nullptr, rb);

    // --- chunk-parallel WKV scan (local -> prefix -> apply) ---
    wkv_local_kernel<<<scang, blk, 0, stream>>>(kb, vb, tdecay, Saa, Sbb, Spp);
    wkv_prefix_kernel<<<D_DIM / 256, blk, 0, stream>>>(tdecay, Saa, Sbb, Spp);
    wkv_apply_kernel<<<scang, blk, 0, stream>>>(kb, vb, rb, tfirst, tdecay,
                                                Saa, Sbb, Spp, kb);

    // rz = x + (wkv*r) @ atto_w^T + atto_b
    gemm_rwkv<0,0,0,1><<<grd, blk, 0, stream>>>(kb, nullptr, w_atto, atto_b, nullptr, x, vb);

    // --- channel-mix branch ---
    ln_kernel<<<T_DIM, blk, 0, stream>>>(vb, ln2_g, ln2_b, xy);                        // ddd
    gemm_rwkv<1,1,0,0><<<grd, blk, 0, stream>>>(xy, ffnmixr, w_ffnr, ffnr_b, nullptr, nullptr, rb); // rf
    gemm_rwkv<1,2,0,0><<<grd, blk, 0, stream>>>(xy, ffnmixk, w_ffnk, ffnk_b, nullptr, nullptr, kb); // relu(kf)^2
    // out = (kf^2 @ ffnv_w^T + ffnv_b) * rf + rz
    gemm_rwkv<0,0,1,1><<<grd, blk, 0, stream>>>(kb, nullptr, w_ffnv, ffnv_b, rb, vb, outp);
}
